// WindowCrossAttention3D_FAST_66099546685508
// MI455X (gfx1250) — matrix operand order repaired and checked by reading, not compiled, not run on hardware
//
#include <hip/hip_runtime.h>

typedef __attribute__((ext_vector_type(16))) _Float16 v16h;
typedef __attribute__((ext_vector_type(8)))  _Float16 v8h;
typedef __attribute__((ext_vector_type(4)))  _Float16 v4h;
typedef __attribute__((ext_vector_type(8)))  float    v8f;
typedef __attribute__((ext_vector_type(4)))  float    v4f;

constexpr int   kDim   = 384;
constexpr int   kHeads = 12;
constexpr int   kHd    = 32;
constexpr int   kSeq   = 512;
constexpr int   kBatch = 64;
constexpr int   kRows  = kBatch * kSeq;        // 32768
constexpr float kScale = 0.17677669529663688f; // 1/sqrt(32)

// ---------------------------------------------------------------------------
// WMMA fragment helpers (wave32, 16x16x32 f16 shapes per CDNA5 ISA layouts)
// A 16x32 (MxK), row-major source with leading dim ld:
//   lane<16 : M=lane,   K in {0..7, 16..23}
//   lane>=16: M=lane-16,K in {8..15,24..31}
__device__ __forceinline__ v16h load_frag_a(const _Float16* base, int ld) {
  const int lane = threadIdx.x & 31;
  const int row  = lane & 15;
  const int kb   = (lane >> 4) << 3;     // 0 or 8
  const _Float16* p = base + row * ld + kb;
  v8h lo = *(const v8h*)(p);
  v8h hi = *(const v8h*)(p + 16);
  v16h a;
#pragma unroll
  for (int i = 0; i < 8; ++i) { a[i] = lo[i]; a[i + 8] = hi[i]; }
  return a;
}

// B 32x16 (KxN) where the source stores rows of B^T: element B(k,n) = src[n*ld + k].
//   lane<16 : N=lane,    K in {0..7, 16..23} ; lane>=16: N=lane-16, K in {8..15, 24..31}
__device__ __forceinline__ v16h load_frag_bt(const _Float16* base, int ld) {
  const int lane = threadIdx.x & 31;
  const int col  = lane & 15;
  const int kb   = (lane >> 4) << 3;     // 0 or 8
  const _Float16* p = base + col * ld + kb;
  v8h lo = *(const v8h*)(p);
  v8h hi = *(const v8h*)(p + 16);
  v16h b;
#pragma unroll
  for (int i = 0; i < 8; ++i) { b[i] = lo[i]; b[i + 8] = hi[i]; }
  return b;
}

__device__ __forceinline__ v8f wmma_f16(v16h a, v16h b, v8f c) {
  return __builtin_amdgcn_wmma_f32_16x16x32_f16(false, a, false, b, (short)0, c,
                                                false, false);
}

// ---------------------------------------------------------------------------
// DPP16 cross-lane reduction over a 16-lane row (both wave32 halves reduce
// independently -- matches the C-fragment column layout). One VALU op/stage.
__device__ __forceinline__ float row16_max(float x) {
  asm("v_max_num_f32_dpp %0, %0, %0 quad_perm:[1,0,3,2] row_mask:0xf bank_mask:0xf\n\t"
      "v_max_num_f32_dpp %0, %0, %0 quad_perm:[2,3,0,1] row_mask:0xf bank_mask:0xf\n\t"
      "v_max_num_f32_dpp %0, %0, %0 row_half_mirror row_mask:0xf bank_mask:0xf\n\t"
      "v_max_num_f32_dpp %0, %0, %0 row_mirror row_mask:0xf bank_mask:0xf"
      : "+v"(x));
  return x;
}
__device__ __forceinline__ float row16_sum(float x) {
  asm("v_add_f32_dpp %0, %0, %0 quad_perm:[1,0,3,2] row_mask:0xf bank_mask:0xf\n\t"
      "v_add_f32_dpp %0, %0, %0 quad_perm:[2,3,0,1] row_mask:0xf bank_mask:0xf\n\t"
      "v_add_f32_dpp %0, %0, %0 row_half_mirror row_mask:0xf bank_mask:0xf\n\t"
      "v_add_f32_dpp %0, %0, %0 row_mirror row_mask:0xf bank_mask:0xf"
      : "+v"(x));
  return x;
}

// ---------------------------------------------------------------------------
// Kernel 0: expand relative-position bias into C-fragment layout:
//   biasf[((h*32 + qi)*32 + ki)*32*8 + lane*8 + v]
// holds bias(h, row = qi*16 + (lane>=16)*8 + v, col = ki*16 + (lane&15)),
// i.e. exactly the f32 16x16 C/D VGPR layout -> fed to WMMA as the C operand.
__global__ void bias_expand_kernel(const float* __restrict__ table,
                                   const int*   __restrict__ relidx,
                                   float*       __restrict__ biasf) {
  long i = (long)blockIdx.x * blockDim.x + threadIdx.x;
  const long total = (long)kHeads * kSeq * kSeq;
  if (i >= total) return;
  int v    = (int)(i & 7);
  int lane = (int)((i >> 3) & 31);
  int ki   = (int)((i >> 8) & 31);
  int qi   = (int)((i >> 13) & 31);
  int h    = (int)(i >> 18);
  int row  = qi * 16 + ((lane >> 4) << 3) + v;
  int col  = ki * 16 + (lane & 15);
  biasf[i] = table[relidx[row * kSeq + col] * kHeads + h];
}

// ---------------------------------------------------------------------------
// Kernels 1/2: projection GEMM, out = X[M,384] @ W[N,384]^T (+bias), f16 out.
// MODE 0: Q  (add q_bias, fold 1/sqrt(hd), out0 = Q[B,H,N,hd])
// MODE 1: KV (cols<384 -> K no bias; cols>=384 -> V + v_bias)
template <int MODE>
__global__ void proj_kernel(const float* __restrict__ X,
                            const float* __restrict__ W,
                            const float* __restrict__ bias,
                            _Float16* __restrict__ out0,
                            _Float16* __restrict__ out1) {
  constexpr int BK = 64;
  extern __shared__ char smem[];
  _Float16* sA = (_Float16*)smem;          // [128][64]
  _Float16* sB = sA + 128 * BK;            // [128][64]

  const int tid  = threadIdx.x;
  const int lane = tid & 31;
  const int wave = tid >> 5;
  const int m0 = blockIdx.x * 128;
  const int n0 = blockIdx.y * 128;
  const int wm = wave & 1;                 // 2 row-bands of 64
  const int wn = wave >> 1;                // 4 col-bands of 32

  v8f acc[4][2] = {};

  for (int kc = 0; kc < kDim; kc += BK) {
    if (kc + BK < kDim)
      __builtin_prefetch(X + (size_t)(m0 + (tid >> 1)) * kDim + kc + BK, 0, 3);
    // cooperative fp32 -> f16 staging: 2 tiles of 128x64
#pragma unroll
    for (int it = 0; it < 8; ++it) {
      int idx = (it * 256 + tid) * 4;
      int r = idx >> 6, c = idx & 63;
      v4f xa = *(const v4f*)(X + (size_t)(m0 + r) * kDim + kc + c);
      v4h ha; ha[0] = (_Float16)xa[0]; ha[1] = (_Float16)xa[1];
              ha[2] = (_Float16)xa[2]; ha[3] = (_Float16)xa[3];
      *(v4h*)(sA + r * BK + c) = ha;
      v4f xb = *(const v4f*)(W + (size_t)(n0 + r) * kDim + kc + c);
      v4h hb; hb[0] = (_Float16)xb[0]; hb[1] = (_Float16)xb[1];
              hb[2] = (_Float16)xb[2]; hb[3] = (_Float16)xb[3];
      *(v4h*)(sB + r * BK + c) = hb;
    }
    __syncthreads();
#pragma unroll
    for (int kk = 0; kk < BK; kk += 32) {
      v16h b0 = load_frag_bt(sB + (wn * 32 +  0) * BK + kk, BK);
      v16h b1 = load_frag_bt(sB + (wn * 32 + 16) * BK + kk, BK);
#pragma unroll
      for (int i = 0; i < 4; ++i) {
        v16h a = load_frag_a(sA + (wm * 64 + i * 16) * BK + kk, BK);
        acc[i][0] = wmma_f16(a, b0, acc[i][0]);
        acc[i][1] = wmma_f16(a, b1, acc[i][1]);
      }
    }
    __syncthreads();
  }

  // epilogue: C/D layout -> lane<16: (M=v, N=lane); lane>=16: (M=v+8, N=lane-16)
  const int cn = lane & 15;
  const int mb = (lane >> 4) << 3;
#pragma unroll
  for (int i = 0; i < 4; ++i) {
#pragma unroll
    for (int j = 0; j < 2; ++j) {
      int col = n0 + wn * 32 + j * 16 + cn;
#pragma unroll
      for (int v = 0; v < 8; ++v) {
        int row = m0 + wm * 64 + i * 16 + mb + v;     // global token row
        int b   = row >> 9, tok = row & 511;
        float val = acc[i][j][v];
        if (MODE == 0) {
          int h = col >> 5, d = col & 31;
          val = (val + bias[col]) * kScale;
          out0[(((size_t)b * kHeads + h) * kSeq + tok) * kHd + d] = (_Float16)val;
        } else {
          if (col < kDim) {
            int h = col >> 5, d = col & 31;
            out0[(((size_t)b * kHeads + h) * kSeq + tok) * kHd + d] = (_Float16)val;
          } else {
            int c2 = col - kDim;
            int h = c2 >> 5, d = c2 & 31;
            val += bias[c2];
            out1[(((size_t)b * kHeads + h) * kSeq + tok) * kHd + d] = (_Float16)val;
          }
        }
      }
    }
  }
}

// ---------------------------------------------------------------------------
// Kernel 3: flash attention, one block per (b,h). 256 threads = 8 waves.
// Each wave owns a 16-query strip; K streamed in 64-column chunks (4 score
// tiles) with online softmax. Bias enters as the WMMA C operand; row
// reductions are single-instruction DPP16 ops.
__global__ void attn_kernel(const _Float16* __restrict__ Qh,
                            const _Float16* __restrict__ Kh,
                            const _Float16* __restrict__ Vh,
                            const float*    __restrict__ biasf,
                            float*          __restrict__ out) {
  extern __shared__ char smem[];
  _Float16* sK  = (_Float16*)smem;          // [512][32]   K row-major
  _Float16* sVt = sK + kSeq * kHd;          // [32][512]   V transposed
  _Float16* sP  = sVt + kHd * kSeq;         // [8][16][64] per-wave P chunk

  const int tid  = threadIdx.x;
  const int lane = tid & 31;
  const int wave = tid >> 5;
  const int bh = blockIdx.x;
  const int b = bh / kHeads, h = bh % kHeads;

  const _Float16* Qb = Qh + (size_t)bh * kSeq * kHd;
  const _Float16* Kb = Kh + (size_t)bh * kSeq * kHd;
  const _Float16* Vb = Vh + (size_t)bh * kSeq * kHd;

  // stage K as-is (vec8) and V transposed
#pragma unroll
  for (int it = 0; it < 8; ++it) {
    int idx = (it * 256 + tid) * 8;
    *(v8h*)(sK + idx) = *(const v8h*)(Kb + idx);
  }
  for (int it = 0; it < 64; ++it) {
    int idx = it * 256 + tid;                 // over 512*32
    int k = idx >> 5, d = idx & 31;
    sVt[d * kSeq + k] = Vb[idx];
  }
  __syncthreads();

  _Float16* myP = sP + (size_t)wave * 16 * 64;
  const int cn = lane & 15;
  const int mb = (lane >> 4) << 3;

  for (int qt = 0; qt < 4; ++qt) {
    const int q0 = (qt * 8 + wave) * 16;
    const int qi = q0 >> 4;
    const v16h qfrag = load_frag_a(Qb + q0 * kHd, kHd);
    // bias tiles for this (h, qi): tile ki at bq + ki*256 floats
    const float* bq = biasf + ((size_t)(h * 32 + qi) * 1024 + lane) * 8;

    v8f O0 = {}, O1 = {};
    float m[8], l[8];
#pragma unroll
    for (int v = 0; v < 8; ++v) { m[v] = -3.0e38f; l[v] = 0.f; }

    for (int c = 0; c < 8; ++c) {             // 8 chunks of 64 columns
      const int k0 = c * 64;
      const float* bp = bq + (size_t)(4 * c) * 256;

      // 4 score tiles: S = Q (16x32) @ K^T (32x16) + bias (as WMMA C operand)
      v8f s[4];
#pragma unroll
      for (int t = 0; t < 4; ++t) {
        v8f cb = *(const v8f*)(bp + t * 256);          // bias in C layout
        v16h kf = load_frag_bt(sK + (k0 + t * 16) * kHd, kHd);
        s[t] = wmma_f16(qfrag, kf, cb);
      }

      // online softmax update (rows live in a fixed 16-lane row)
      float alpha[8];
#pragma unroll
      for (int v = 0; v < 8; ++v) {
        float tm = fmaxf(fmaxf(s[0][v], s[1][v]), fmaxf(s[2][v], s[3][v]));
        tm = row16_max(tm);
        float mn = fmaxf(m[v], tm);
        alpha[v] = __expf(m[v] - mn);
        m[v] = mn;
      }
#pragma unroll
      for (int v = 0; v < 8; ++v) {
        float ts = 0.f;
#pragma unroll
        for (int t = 0; t < 4; ++t) {
          s[t][v] = __expf(s[t][v] - m[v]);
          ts += s[t][v];
        }
        ts = row16_sum(ts);
        l[v] = l[v] * alpha[v] + ts;
        O0[v] *= alpha[v];
        O1[v] *= alpha[v];
      }

      // P chunk (16x64 f16) to per-wave LDS strip, then O += P @ V chunk
#pragma unroll
      for (int t = 0; t < 4; ++t)
#pragma unroll
        for (int v = 0; v < 8; ++v)
          myP[(mb + v) * 64 + t * 16 + cn] = (_Float16)s[t][v];
      asm volatile("s_wait_dscnt 0" ::: "memory");

      v16h pa0  = load_frag_a(myP, 64);            // P cols  0..31
      v16h pa1  = load_frag_a(myP + 32, 64);       // P cols 32..63
      v16h vb00 = load_frag_bt(sVt + 0  * kSeq + k0,      kSeq);
      v16h vb01 = load_frag_bt(sVt + 16 * kSeq + k0,      kSeq);
      v16h vb10 = load_frag_bt(sVt + 0  * kSeq + k0 + 32, kSeq);
      v16h vb11 = load_frag_bt(sVt + 16 * kSeq + k0 + 32, kSeq);
      O0 = wmma_f16(pa0, vb00, O0);
      O1 = wmma_f16(pa0, vb01, O1);
      O0 = wmma_f16(pa1, vb10, O0);
      O1 = wmma_f16(pa1, vb11, O1);
    }

    // ---- normalize and store fp32 out[b, tok, h*32 + d]
#pragma unroll
    for (int v = 0; v < 8; ++v) {
      int row = q0 + mb + v;
      float inv = 1.f / l[v];
      size_t base = ((size_t)b * kSeq + row) * kDim + h * kHd;
      out[base +      cn] = O0[v] * inv;
      out[base + 16 + cn] = O1[v] * inv;
    }
  }
}

// ---------------------------------------------------------------------------
extern "C" void kernel_launch(void* const* d_in, const int* in_sizes, int n_in,
                              void* d_out, int out_size, void* d_ws, size_t ws_size,
                              hipStream_t stream) {
  const float* x1         = (const float*)d_in[0];
  const float* x2         = (const float*)d_in[1];
  const float* Wq         = (const float*)d_in[2];
  const float* Wkv        = (const float*)d_in[3];
  const float* q_bias     = (const float*)d_in[4];
  const float* v_bias     = (const float*)d_in[5];
  const float* bias_table = (const float*)d_in[6];
  const int*   rel_index  = (const int*)d_in[7];
  float* out = (float*)d_out;

  const size_t qkv_elems = (size_t)kBatch * kHeads * kSeq * kHd; // 12.58M
  _Float16* Qh = (_Float16*)d_ws;
  _Float16* Kh = Qh + qkv_elems;
  _Float16* Vh = Kh + qkv_elems;
  float* biasf = (float*)(Vh + qkv_elems);                       // [12,32,32,32,8]

  dim3 blk(256);

  // 0) bias expand into fragment layout: 12*512*512 elements
  {
    long total = (long)kHeads * kSeq * kSeq;
    int blocks = (int)((total + 255) / 256);
    bias_expand_kernel<<<blocks, blk, 0, stream>>>(bias_table, rel_index, biasf);
  }

  // 1) Q projection: 32768 x 384
  {
    dim3 grid(kRows / 128, kDim / 128);
    size_t lds = 2 * 128 * 64 * sizeof(_Float16);
    proj_kernel<0><<<grid, blk, lds, stream>>>(x1, Wq, q_bias, Qh, nullptr);
  }

  // 2) KV projection: 32768 x 768
  {
    dim3 grid(kRows / 128, (2 * kDim) / 128);
    size_t lds = 2 * 128 * 64 * sizeof(_Float16);
    proj_kernel<1><<<grid, blk, lds, stream>>>(x2, Wkv, v_bias, Kh, Vh);
  }

  // 3) attention: one block per (b, h)
  {
    size_t lds = (size_t)(kSeq * kHd + kHd * kSeq + 8 * 16 * 64) * sizeof(_Float16);
    attn_kernel<<<kBatch * kHeads, blk, lds, stream>>>(Qh, Kh, Vh, biasf, out);
  }
}
